// GKCIModel_12506944766111
// MI455X (gfx1250) — matrix operand order repaired and checked by reading, not compiled, not run on hardware
//
#include <hip/hip_runtime.h>
#include <cstddef>

#define D_EMB 128

typedef __attribute__((ext_vector_type(16))) _Float16 v16h;
typedef __attribute__((ext_vector_type(8)))  _Float16 v8h;
typedef __attribute__((ext_vector_type(8)))  float    v8f;

__device__ __forceinline__ v16h make_frag(const _Float16* p0, const _Float16* p1) {
    v8h lo = *(const v8h*)p0;
    v8h hi = *(const v8h*)p1;
    v16h r;
#pragma unroll
    for (int i = 0; i < 8; ++i) { r[i] = lo[i]; r[i + 8] = hi[i]; }
    return r;
}

__device__ __forceinline__ float sigmoidf_(float x) { return 1.0f / (1.0f + __expf(-x)); }

// order-preserving float <-> uint encoding for atomic max on signed floats
__device__ __forceinline__ unsigned ord_enc(float f) {
    unsigned u = __float_as_uint(f);
    return (u & 0x80000000u) ? ~u : (u | 0x80000000u);
}
__device__ __forceinline__ float ord_dec(unsigned u) {
    unsigned b = (u & 0x80000000u) ? (u ^ 0x80000000u) : ~u;
    return __uint_as_float(b);
}

__device__ __forceinline__ void atomic_add_f32(float* p, float v) {
    unsafeAtomicAdd(p, v);   // lowers to global_atomic_add_f32 (no CAS loop)
}

// ---------------------------------------------------------------------------
// ScoringNet: Linear(128,64)+BN+LReLU(0.2) -> Linear(64,32)+BN+LReLU(0.2)
//             -> Linear(32,1), all fused, WMMA f16 inputs / f32 accum.
// Block = 128 threads = 4 waves; each wave owns a 16-node M-tile.
// ---------------------------------------------------------------------------
__global__ __launch_bounds__(128)
void scoring_kernel(const float* __restrict__ emb,
                    const float* __restrict__ w1, const float* __restrict__ b1,
                    const float* __restrict__ g1, const float* __restrict__ bb1,
                    const float* __restrict__ w2, const float* __restrict__ b2,
                    const float* __restrict__ g2, const float* __restrict__ bb2,
                    const float* __restrict__ w3, const float* __restrict__ b3,
                    float* __restrict__ scores, float* __restrict__ original, int N)
{
    __shared__ _Float16 W1h[64 * 128];   // 16 KB
    __shared__ _Float16 W2h[32 * 64];    //  4 KB
    __shared__ _Float16 At[4][16 * 128]; // 16 KB (per-wave A tiles)
    __shared__ _Float16 H1[4][16 * 64];  //  8 KB
    __shared__ float    H2[4][16 * 32];  //  8 KB

    const int t    = threadIdx.x;
    const int w    = t >> 5;
    const int lane = t & 31;
    const int half = lane >> 4;
    const int lq   = lane & 15;
    const int blockBase = blockIdx.x * 64;
    const float bninv = rsqrtf(1.0f + 1e-5f);

    // stage W1 (64x128) as f16: 64 elems/thread
    {
        const float4* src = (const float4*)w1;
#pragma unroll
        for (int i = 0; i < 16; ++i) {
            float4 v = src[t * 16 + i];
            int o = t * 64 + i * 4;
            W1h[o + 0] = (_Float16)v.x; W1h[o + 1] = (_Float16)v.y;
            W1h[o + 2] = (_Float16)v.z; W1h[o + 3] = (_Float16)v.w;
        }
    }
    // stage W2 (32x64) as f16: 16 elems/thread
    {
        const float4* src = (const float4*)w2;
#pragma unroll
        for (int i = 0; i < 4; ++i) {
            float4 v = src[t * 4 + i];
            int o = t * 16 + i * 4;
            W2h[o + 0] = (_Float16)v.x; W2h[o + 1] = (_Float16)v.y;
            W2h[o + 2] = (_Float16)v.z; W2h[o + 3] = (_Float16)v.w;
        }
    }
    // stage this wave's 16 embedding rows (clamped) as f16
    {
        int node = blockBase + w * 16 + lq;
        if (node >= N) node = N - 1;
        const float4* src = (const float4*)(emb + (size_t)node * D_EMB + half * 64);
        int o = lq * 128 + half * 64;
#pragma unroll
        for (int i = 0; i < 16; ++i) {
            float4 v = src[i];
            At[w][o + i * 4 + 0] = (_Float16)v.x; At[w][o + i * 4 + 1] = (_Float16)v.y;
            At[w][o + i * 4 + 2] = (_Float16)v.z; At[w][o + i * 4 + 3] = (_Float16)v.w;
        }
    }
    __syncthreads();

    // layer 1: [16x64] = [16x128] @ W1^T  -> 4 n-tiles x 4 k-steps
#pragma unroll
    for (int nt = 0; nt < 4; ++nt) {
        const int f = nt * 16 + lq;
        const float bias = b1[f];
        v8f acc;
#pragma unroll
        for (int r = 0; r < 8; ++r) acc[r] = bias;
#pragma unroll
        for (int kt = 0; kt < 4; ++kt) {
            const _Float16* ap = &At[w][lq * 128 + kt * 32 + half * 8];
            v16h a = make_frag(ap, ap + 16);
            const _Float16* bp = &W1h[f * 128 + kt * 32 + half * 8];
            v16h bf = make_frag(bp, bp + 16);
            acc = __builtin_amdgcn_wmma_f32_16x16x32_f16(false, a, false, bf,
                                                         (short)0, acc, false, false);
        }
        const float gs = g1[f] * bninv, bo = bb1[f];
#pragma unroll
        for (int r = 0; r < 8; ++r) {
            float v = acc[r] * gs + bo;
            v = (v >= 0.0f) ? v : 0.2f * v;
            H1[w][(r + 8 * half) * 64 + f] = (_Float16)v;
        }
    }
    __syncthreads();

    // layer 2: [16x32] = [16x64] @ W2^T  -> 2 n-tiles x 2 k-steps
#pragma unroll
    for (int nt = 0; nt < 2; ++nt) {
        const int f = nt * 16 + lq;
        const float bias = b2[f];
        v8f acc;
#pragma unroll
        for (int r = 0; r < 8; ++r) acc[r] = bias;
#pragma unroll
        for (int kt = 0; kt < 2; ++kt) {
            const _Float16* ap = &H1[w][lq * 64 + kt * 32 + half * 8];
            v16h a = make_frag(ap, ap + 16);
            const _Float16* bp = &W2h[f * 64 + kt * 32 + half * 8];
            v16h bf = make_frag(bp, bp + 16);
            acc = __builtin_amdgcn_wmma_f32_16x16x32_f16(false, a, false, bf,
                                                         (short)0, acc, false, false);
        }
        const float gs = g2[f] * bninv, bo = bb2[f];
#pragma unroll
        for (int r = 0; r < 8; ++r) {
            float v = acc[r] * gs + bo;
            v = (v >= 0.0f) ? v : 0.2f * v;
            H2[w][(r + 8 * half) * 32 + f] = v;
        }
    }
    __syncthreads();

    // layer 3: score[m] = H2[m,:] . w3 + b3  (16 lanes per wave)
    if (half == 0) {
        int node = blockBase + w * 16 + lq;
        if (node < N) {
            float s = b3[0];
#pragma unroll
            for (int k = 0; k < 32; ++k) s += H2[w][lq * 32 + k] * w3[k];
            scores[node]   = s;
            original[node] = s;
        }
    }
}

// ---------------------------------------------------------------------------
// Edge-aggregation passes (memory/atomic bound).
// Fused form: softmax-agg = (sum_e e*w*s) / (sum_e e)  -> only 2 edge passes.
// ---------------------------------------------------------------------------
__global__ void init_kernel(unsigned* maxIn, unsigned* maxOut,
                            float* denomIn, float* denomOut,
                            float* numIn, float* numOut,
                            float* deg, int N, int zeroDeg)
{
    int i = blockIdx.x * blockDim.x + threadIdx.x;
    if (i < N) {
        maxIn[i] = 0u; maxOut[i] = 0u;
        denomIn[i] = 0.0f; denomOut[i] = 0.0f;
        numIn[i] = 0.0f;   numOut[i] = 0.0f;
        if (zeroDeg) deg[i] = 0.0f;
    }
}

__global__ void edge_max_kernel(const float* __restrict__ scores,
                                const int* __restrict__ src, const int* __restrict__ dst,
                                unsigned* maxIn, unsigned* maxOut, float* deg,
                                int E, int addDeg)
{
    int e = blockIdx.x * blockDim.x + threadIdx.x;
    if (e < E) {
        int u = src[e], v = dst[e];
        float su = scores[u], sv = scores[v];
        __hip_atomic_fetch_max(&maxIn[v],  ord_enc(su), __ATOMIC_RELAXED, __HIP_MEMORY_SCOPE_AGENT);
        __hip_atomic_fetch_max(&maxOut[u], ord_enc(sv), __ATOMIC_RELAXED, __HIP_MEMORY_SCOPE_AGENT);
        if (addDeg) atomic_add_f32(&deg[v], 1.0f);
    }
}

__global__ void edge_sum_kernel(const float* __restrict__ scores,
                                const int* __restrict__ src, const int* __restrict__ dst,
                                const float* __restrict__ ew,
                                const unsigned* __restrict__ maxIn,
                                const unsigned* __restrict__ maxOut,
                                float* denomIn, float* denomOut,
                                float* numIn, float* numOut, int E)
{
    int e = blockIdx.x * blockDim.x + threadIdx.x;
    if (e < E) {
        int u = src[e], v = dst[e];
        float su = scores[u], sv = scores[v];
        float wgt = ew[e];
        float ein  = __expf(su - ord_dec(maxIn[v]));
        float eout = __expf(sv - ord_dec(maxOut[u]));
        atomic_add_f32(&denomIn[v],  ein);
        atomic_add_f32(&numIn[v],    ein * wgt * su);
        atomic_add_f32(&denomOut[u], eout);
        atomic_add_f32(&numOut[u],   eout * wgt * sv);
    }
}

// ---------------------------------------------------------------------------
// Per-node: finish softmax division, update MLP (3->24->12->1, sigmoid), blend
// ---------------------------------------------------------------------------
__global__ void update_kernel(float* scores, const float* __restrict__ original,
                              const float* __restrict__ denomIn, const float* __restrict__ denomOut,
                              const float* __restrict__ numIn, const float* __restrict__ numOut,
                              const float* __restrict__ w1, const float* __restrict__ b1,
                              const float* __restrict__ g1, const float* __restrict__ bb1,
                              const float* __restrict__ w2, const float* __restrict__ b2,
                              const float* __restrict__ w3, const float* __restrict__ b3,
                              float alpha, int N)
{
    int i = blockIdx.x * blockDim.x + threadIdx.x;
    if (i >= N) return;
    const float bninv = rsqrtf(1.0f + 1e-5f);
    float x0 = scores[i];
    float dIn = denomIn[i], dOut = denomOut[i];
    float x1 = (dIn  > 0.0f) ? numIn[i]  / dIn  : 0.0f;   // agg over predecessors
    float x2 = (dOut > 0.0f) ? numOut[i] / dOut : 0.0f;   // agg over successors
    float h1[24];
#pragma unroll
    for (int j = 0; j < 24; ++j) {
        float h = w1[j * 3 + 0] * x0 + w1[j * 3 + 1] * x1 + w1[j * 3 + 2] * x2 + b1[j];
        h = h * (g1[j] * bninv) + bb1[j];
        h1[j] = (h >= 0.0f) ? h : 0.1f * h;
    }
    float h2[12];
#pragma unroll
    for (int j = 0; j < 12; ++j) {
        float h = b2[j];
#pragma unroll
        for (int k = 0; k < 24; ++k) h += w2[j * 24 + k] * h1[k];
        h2[j] = (h >= 0.0f) ? h : 0.1f * h;
    }
    float o = b3[0];
#pragma unroll
    for (int k = 0; k < 12; ++k) o += w3[k] * h2[k];
    o = sigmoidf_(o);
    scores[i] = alpha * o + (1.0f - alpha) * original[i];
}

// ---------------------------------------------------------------------------
// Final centrality / mixing / noise-gate elementwise stage
// ---------------------------------------------------------------------------
__global__ void final_kernel(const float* __restrict__ scores,
                             const float* __restrict__ original,
                             const float* __restrict__ deg,
                             const float* __restrict__ gamma, const float* __restrict__ beta,
                             const float* __restrict__ cscale,
                             const float* __restrict__ mwraw, const float* __restrict__ ntr,
                             float* __restrict__ out, int N)
{
    int i = blockIdx.x * blockDim.x + threadIdx.x;
    if (i >= N) return;
    float cv  = __logf(deg[i] * cscale[0] + 1e-6f);
    float cst = gamma[0] * cv + beta[0];
    float sm  = 5.0f * tanhf(cst * 0.2f);
    float fs  = sigmoidf_(sm * scores[i]);
    float mw  = sigmoidf_(mwraw[0]);
    float mixed = mw * fs + (1.0f - mw) * sigmoidf_(original[i]);
    float thr = sigmoidf_(ntr[0]);
    if (mixed < thr) mixed *= 0.5f;
    out[i] = fminf(fmaxf(mixed, 0.0f), 1.0f);
}

// ---------------------------------------------------------------------------
extern "C" void kernel_launch(void* const* d_in, const int* in_sizes, int n_in,
                              void* d_out, int out_size, void* d_ws, size_t ws_size,
                              hipStream_t stream)
{
    (void)n_in; (void)ws_size;
    // --- detect flattening convention (insertion-order vs jax alphabetical) ---
    int iEmb, iSrc, iDst, iW;
    if (in_sizes[0] > 4000000) { iEmb = 0; iSrc = 1; iDst = 2; iW = 3; }      // insertion
    else                       { iDst = 0; iSrc = 1; iW = 2; iEmb = 3; }      // alphabetical

    const int p = 4;
    int iSW1,iSB1,iSG1,iSBB1,iSW2,iSB2,iSG2,iSBB2,iSW3,iSB3;
    int iGamma,iBeta,iCS,iMW,iNT;
    int iU[2][8]; // [l]{w1,b1,g1,bb1,w2,b2,w3,b3}
    if (in_sizes[p] == 8192) { // insertion-order params
        iSW1=p+0; iSB1=p+1; iSG1=p+2; iSBB1=p+3; iSW2=p+4; iSB2=p+5; iSG2=p+6; iSBB2=p+7;
        iSW3=p+8; iSB3=p+9;
        iGamma=p+10; iBeta=p+11; iCS=p+12; iMW=p+13; iNT=p+14;
        for (int l = 0; l < 2; ++l) {
            int b = p + 15 + 8 * l;
            iU[l][0]=b+0; iU[l][1]=b+1; iU[l][2]=b+2; iU[l][3]=b+3;
            iU[l][4]=b+4; iU[l][5]=b+5; iU[l][6]=b+6; iU[l][7]=b+7;
        }
    } else {                   // alphabetical params
        iBeta=p+0; iCS=p+1; iGamma=p+2; iMW=p+3; iNT=p+4;
        iSB1=p+5; iSB2=p+6; iSB3=p+7; iSBB1=p+8; iSBB2=p+9;
        iSG1=p+10; iSG2=p+11; iSW1=p+12; iSW2=p+13; iSW3=p+14;
        for (int l = 0; l < 2; ++l) {
            int b = p + 15 + 8 * l; // order: b1,b2,b3,bb1,g1,w1,w2,w3
            iU[l][1]=b+0; iU[l][5]=b+1; iU[l][7]=b+2; iU[l][3]=b+3;
            iU[l][2]=b+4; iU[l][0]=b+5; iU[l][4]=b+6; iU[l][6]=b+7;
        }
    }

    const int N = in_sizes[iEmb] / D_EMB;
    const int E = in_sizes[iSrc];

    const float* emb = (const float*)d_in[iEmb];
    const int*   src = (const int*)d_in[iSrc];
    const int*   dst = (const int*)d_in[iDst];
    const float* ew  = (const float*)d_in[iW];

    float* ws       = (float*)d_ws;
    float* scores   = ws;
    float* original = ws + (size_t)N;
    unsigned* maxIn  = (unsigned*)(ws + 2 * (size_t)N);
    unsigned* maxOut = (unsigned*)(ws + 3 * (size_t)N);
    float* denomIn  = ws + 4 * (size_t)N;
    float* denomOut = ws + 5 * (size_t)N;
    float* numIn    = ws + 6 * (size_t)N;
    float* numOut   = ws + 7 * (size_t)N;
    float* deg      = ws + 8 * (size_t)N;

    const int TB = 256;
    dim3 gN((N + TB - 1) / TB), gE((E + TB - 1) / TB), bT(TB);

    scoring_kernel<<<(N + 63) / 64, 128, 0, stream>>>(
        emb,
        (const float*)d_in[iSW1], (const float*)d_in[iSB1],
        (const float*)d_in[iSG1], (const float*)d_in[iSBB1],
        (const float*)d_in[iSW2], (const float*)d_in[iSB2],
        (const float*)d_in[iSG2], (const float*)d_in[iSBB2],
        (const float*)d_in[iSW3], (const float*)d_in[iSB3],
        scores, original, N);

    for (int l = 0; l < 2; ++l) {
        float alpha = 0.7f + 0.2f * (float)(l + 1) / 2.0f;
        init_kernel<<<gN, bT, 0, stream>>>(maxIn, maxOut, denomIn, denomOut,
                                           numIn, numOut, deg, N, l == 0);
        edge_max_kernel<<<gE, bT, 0, stream>>>(scores, src, dst, maxIn, maxOut,
                                               deg, E, l == 0);
        edge_sum_kernel<<<gE, bT, 0, stream>>>(scores, src, dst, ew, maxIn, maxOut,
                                               denomIn, denomOut, numIn, numOut, E);
        update_kernel<<<gN, bT, 0, stream>>>(
            scores, original, denomIn, denomOut, numIn, numOut,
            (const float*)d_in[iU[l][0]], (const float*)d_in[iU[l][1]],
            (const float*)d_in[iU[l][2]], (const float*)d_in[iU[l][3]],
            (const float*)d_in[iU[l][4]], (const float*)d_in[iU[l][5]],
            (const float*)d_in[iU[l][6]], (const float*)d_in[iU[l][7]],
            alpha, N);
    }

    final_kernel<<<gN, bT, 0, stream>>>(scores, original, deg,
                                        (const float*)d_in[iGamma], (const float*)d_in[iBeta],
                                        (const float*)d_in[iCS], (const float*)d_in[iMW],
                                        (const float*)d_in[iNT],
                                        (float*)d_out, N);
    (void)out_size;
}
